// FrequencyMixingTABBlock_66975720014060
// MI455X (gfx1250) — compile-verified, matrix operand order checked by reading, not run on hardware
//
#include <hip/hip_runtime.h>
#include <hip/hip_bf16.h>
#include <cstdint>

// ---------------------------------------------------------------------------
// FrequencyMixingTABBlock for MI455X (gfx1250, wave32, WMMA)
//   N=128, parts=2, c=256, H=W=19.  M = N*H*W = 46208, K = 2c = 512.
//   Channel mixes run as bf16 WMMA GEMMs (v_wmma_f32_16x16x32_bf16),
//   everything else is fp32 bandwidth-bound elementwise / depthwise work.
// ---------------------------------------------------------------------------

typedef __attribute__((ext_vector_type(16))) __bf16 v16bf;
typedef __attribute__((ext_vector_type(8)))  float  v8f;

#define NB   128
#define CCH  256
#define HWW  361            // 19*19
#define PER  (CCH*HWW)      // 92416 per (n, part-plane of one channel-set)
#define MTOT (NB*HWW)       // 46208 (divisible by 16 -> 2888 M-tiles)
#define KTOT (2*CCH)        // 512

__device__ __forceinline__ unsigned short f32_to_bf16(float f) {
  union { float f; unsigned u; } c; c.f = f;
  unsigned u = c.u;
  unsigned r = u + 0x7FFFu + ((u >> 16) & 1u);   // round-to-nearest-even
  return (unsigned short)(r >> 16);
}

__device__ __forceinline__ float silu(float x) {
  return x / (1.0f + __expf(-x));
}

// Load a 16-element bf16 fragment per CDNA5 16-bit A/B layout:
// vector elems 0..7  <- p[0..7]   (VGPR 0..3, K = base .. base+7)
// vector elems 8..15 <- p[16..23] (VGPR 4..7, K = base+16 .. base+23)
__device__ __forceinline__ v16bf load_frag16(const unsigned short* p) {
  union { unsigned int w[8]; v16bf v; } u;
  uint4 a  = *reinterpret_cast<const uint4*>(p);
  uint4 b2 = *reinterpret_cast<const uint4*>(p + 16);
  u.w[0] = a.x;  u.w[1] = a.y;  u.w[2] = a.z;  u.w[3] = a.w;
  u.w[4] = b2.x; u.w[5] = b2.y; u.w[6] = b2.z; u.w[7] = b2.w;
  return u.v;
}

// ---------------- weight conversion fp32 -> bf16 ---------------------------
__global__ __launch_bounds__(256) void cvt_kernel(const float* __restrict__ src,
                                                  unsigned short* __restrict__ dst,
                                                  int n) {
  int i = blockIdx.x * 256 + threadIdx.x;
  if (i < n) dst[i] = f32_to_bf16(src[i]);
}

// ---------------- pre: scale + SiLU + rotate(+angle) -----------------------
__global__ __launch_bounds__(256) void pre_kernel(const float* __restrict__ z,
                                                  const float* __restrict__ cosA,
                                                  const float* __restrict__ sinA,
                                                  const int* __restrict__ bidx,
                                                  float* __restrict__ X) {
  int i = blockIdx.x * 256 + threadIdx.x;
  if (i >= NB * PER) return;
  float inv = rsqrtf((float)(*bidx) + 1.0f);
  int n = i / PER; int j = i - n * PER;
  size_t br = (size_t)n * 2 * PER + j;
  float zr = z[br] * inv;
  float zi = z[br + PER] * inv;
  zr = silu(zr);
  zi = silu(zi);
  float c = cosA[i], s = sinA[i];
  X[br]       = zr * c - zi * s;
  X[br + PER] = zr * s + zi * c;
}

// ---------------- complex depthwise 3x3 + rotate(-angle) -------------------
// One block per (n, group of 8 channels). Stages 16 padded planes (re/im x 8ch)
// in LDS, computes the 4 real convs per tap, rotates by -angle, then does an
// LDS channel transpose so the bf16 GEMM operand lands in (m,k) layout as
// contiguous 16B chunks (A[m*512 + part*256 + ch]).
__global__ __launch_bounds__(256) void conv_kernel(const float* __restrict__ X,
                                                   const float* __restrict__ cosA,
                                                   const float* __restrict__ sinA,
                                                   const float* __restrict__ kr,
                                                   const float* __restrict__ ki,
                                                   unsigned short* __restrict__ A,
                                                   int dil) {
  __shared__ float sm[16 * 529];   // 16 input planes, padded up to 23x23
  __shared__ float so[16 * 368];   // 16 output planes (361, padded stride)
  __shared__ float skr[72], ski[72];

  int tid = threadIdx.x;
  int n   = blockIdx.x >> 5;
  int ch0 = (blockIdx.x & 31) * 8;
  int SD  = 19 + 2 * dil;          // padded side: 21 (dil=1) or 23 (dil=2)
  int SA  = SD * SD;

  if (tid < 72) { skr[tid] = kr[ch0 * 9 + tid]; ski[tid] = ki[ch0 * 9 + tid]; }

  for (int t = tid; t < 16 * SA; t += 256) {
    int p = t / SA; int q = t - p * SA;
    int py = q / SD; int px = q - py * SD;
    int y = py - dil, x = px - dil;
    float v = 0.0f;
    if ((unsigned)y < 19u && (unsigned)x < 19u) {
      int part = p >> 3, cl = p & 7;
      v = X[(size_t)((n * 2 + part) * CCH + ch0 + cl) * HWW + y * 19 + x];
    }
    sm[p * 529 + q] = v;
  }
  __syncthreads();

  for (int t = tid; t < 8 * HWW; t += 256) {
    int cl = t / HWW; int hw = t - cl * HWW;
    int y = hw / 19, x = hw - y * 19;
    float srr = 0.f, sii = 0.f, sri = 0.f, sir = 0.f;
#pragma unroll
    for (int ty = 0; ty < 3; ++ty)
#pragma unroll
      for (int tx = 0; tx < 3; ++tx) {
        int off = (y + dil + (ty - 1) * dil) * SD + (x + dil + (tx - 1) * dil);
        float wr = skr[cl * 9 + ty * 3 + tx];
        float wi = ski[cl * 9 + ty * 3 + tx];
        float re = sm[cl * 529 + off];
        float im = sm[(8 + cl) * 529 + off];
        srr = fmaf(re, wr, srr);
        sii = fmaf(im, wi, sii);
        sri = fmaf(re, wi, sri);
        sir = fmaf(im, wr, sir);
      }
    float orr = srr - sii;          // conv output, real
    float oii = sri + sir;          // conv output, imag
    size_t ci = (size_t)(n * CCH + ch0 + cl) * HWW + hw;
    float c = cosA[ci], s = sinA[ci];
    so[cl * 368 + hw]       = orr * c + oii * s;   // rotate by -angle
    so[(8 + cl) * 368 + hw] = oii * c - orr * s;
  }
  __syncthreads();

  for (int t = tid; t < 2 * HWW; t += 256) {
    int hw = t >> 1; int half = t & 1;   // half 0 = real part (k=ch), 1 = imag (k=256+ch)
    unsigned int w[4];
#pragma unroll
    for (int d = 0; d < 4; ++d) {
      unsigned lo = f32_to_bf16(so[(half * 8 + 2 * d)     * 368 + hw]);
      unsigned hi = f32_to_bf16(so[(half * 8 + 2 * d + 1) * 368 + hw]);
      w[d] = lo | (hi << 16);
    }
    size_t m = (size_t)n * HWW + hw;
    uint4* dst = reinterpret_cast<uint4*>(A + m * KTOT + half * CCH + ch0);
    *dst = make_uint4(w[0], w[1], w[2], w[3]);
  }
}

// ---------------- channel mix: bf16 WMMA GEMM + fused epilogue -------------
// out[m, o] = sum_k A[m,k] * W[o,k].  Each wave: 16 M-rows x 64 channels,
// computing re (o=ch) and im (o=ch+256) tiles together -> 8 accumulators,
// one A fragment amortized over 8 v_wmma per K-step.
// mode 0: epilogue = SiLU + rotate(+angle), write NCHW fp32 (feeds conv2)
// mode 1: epilogue = + residual z, write NCHW fp32 (final output)
__global__ __launch_bounds__(256) void mix_kernel(const unsigned short* __restrict__ A,
                                                  const unsigned short* __restrict__ Wb,
                                                  const float* __restrict__ cosA,
                                                  const float* __restrict__ sinA,
                                                  const float* __restrict__ zskip,
                                                  float* __restrict__ Out,
                                                  int mode) {
  int wid   = blockIdx.x * 8 + (threadIdx.x >> 5);
  int lane  = threadIdx.x & 31;
  int mtile = wid >> 2;          // 0..2887
  int chgrp = wid & 3;           // 64 channels per group
  int lrow  = lane & 15;
  int lhi   = lane >> 4;

  const unsigned short* Arow = A + (size_t)(mtile * 16 + lrow) * KTOT;

  v8f zero = {0.f, 0.f, 0.f, 0.f, 0.f, 0.f, 0.f, 0.f};
  v8f acc[8];
#pragma unroll
  for (int i = 0; i < 8; ++i) acc[i] = zero;

  for (int kk = 0; kk < 16; ++kk) {
    int k0 = kk * 32 + 8 * lhi;
    v16bf a = load_frag16(Arow + k0);
    if (kk < 14) __builtin_prefetch(Arow + k0 + 64, 0, 0);  // global_prefetch_b8
#pragma unroll
    for (int t = 0; t < 4; ++t) {
      int ore = chgrp * 64 + t * 16 + lrow;      // real-part output row
      v16bf br = load_frag16(Wb + (size_t)ore * KTOT + k0);
      v16bf bi = load_frag16(Wb + (size_t)(ore + 256) * KTOT + k0);
      acc[2 * t] = __builtin_amdgcn_wmma_f32_16x16x32_bf16(
          false, a, false, br, (short)0, acc[2 * t], false, false);
      acc[2 * t + 1] = __builtin_amdgcn_wmma_f32_16x16x32_bf16(
          false, a, false, bi, (short)0, acc[2 * t + 1], false, false);
    }
  }

  // C/D layout: lane -> N = lane%16 (channel), VGPR r -> M = r + 8*(lane/16)
#pragma unroll
  for (int t = 0; t < 4; ++t) {
    int ch = chgrp * 64 + t * 16 + lrow;
#pragma unroll
    for (int r = 0; r < 8; ++r) {
      int m = mtile * 16 + r + 8 * lhi;
      int n = m / HWW; int hw = m - n * HWW;
      float vr = acc[2 * t][r];
      float vi = acc[2 * t + 1][r];
      size_t ore = (size_t)((n * 2) * CCH + ch) * HWW + hw;
      size_t oim = ore + (size_t)CCH * HWW;
      if (mode == 0) {
        size_t ci = (size_t)(n * CCH + ch) * HWW + hw;
        float c = cosA[ci], s = sinA[ci];
        vr = silu(vr);
        vi = silu(vi);
        Out[ore] = vr * c - vi * s;   // rotate by +angle
        Out[oim] = vr * s + vi * c;
      } else {
        Out[ore] = vr + zskip[ore];
        Out[oim] = vi + zskip[oim];
      }
    }
  }
}

// ---------------------------------------------------------------------------
extern "C" void kernel_launch(void* const* d_in, const int* in_sizes, int n_in,
                              void* d_out, int out_size, void* d_ws, size_t ws_size,
                              hipStream_t stream) {
  const float* z      = (const float*)d_in[0];
  const float* cosA   = (const float*)d_in[1];
  const float* sinA   = (const float*)d_in[2];
  const int*   bidx   = (const int*)d_in[3];
  const float* dw1_kr = (const float*)d_in[4];
  const float* dw1_ki = (const float*)d_in[5];
  const float* mix1_w = (const float*)d_in[6];
  const float* dw2_kr = (const float*)d_in[7];
  const float* dw2_ki = (const float*)d_in[8];
  const float* mix2_w = (const float*)d_in[9];
  float* out = (float*)d_out;

  // Workspace layout (intermediates stay L2-resident: 94.6MB + 47.3MB + 1MB)
  char* ws = (char*)d_ws;
  float* X = (float*)ws;                                        // (N,2,c,H,W) fp32
  size_t offA = (((size_t)NB * 2 * CCH * HWW * 4) + 255) & ~(size_t)255;
  unsigned short* Abuf = (unsigned short*)(ws + offA);          // (M, K) bf16
  size_t offW = offA + ((((size_t)MTOT * KTOT * 2) + 255) & ~(size_t)255);
  unsigned short* Wb1 = (unsigned short*)(ws + offW);           // 512x512 bf16
  unsigned short* Wb2 = Wb1 + 512 * 512;

  cvt_kernel<<<(512 * 512 + 255) / 256, 256, 0, stream>>>(mix1_w, Wb1, 512 * 512);
  cvt_kernel<<<(512 * 512 + 255) / 256, 256, 0, stream>>>(mix2_w, Wb2, 512 * 512);

  pre_kernel<<<(NB * PER + 255) / 256, 256, 0, stream>>>(z, cosA, sinA, bidx, X);

  conv_kernel<<<NB * 32, 256, 0, stream>>>(X, cosA, sinA, dw1_kr, dw1_ki, Abuf, 2);
  mix_kernel<<<(MTOT / 16) * 4 / 8, 256, 0, stream>>>(Abuf, Wb1, cosA, sinA,
                                                      (const float*)nullptr, X, 0);
  conv_kernel<<<NB * 32, 256, 0, stream>>>(X, cosA, sinA, dw2_kr, dw2_ki, Abuf, 1);
  mix_kernel<<<(MTOT / 16) * 4 / 8, 256, 0, stream>>>(Abuf, Wb2, cosA, sinA, z, out, 1);
}
